// Decoder_12292196401549
// MI455X (gfx1250) — matrix operand order repaired and checked by reading, compiled, not run on hardware
//
#include <hip/hip_runtime.h>
#include <hip/hip_bf16.h>

typedef __attribute__((ext_vector_type(16))) __bf16 v16bf;
typedef __attribute__((ext_vector_type(8)))  __bf16 v8bf;
typedef __attribute__((ext_vector_type(8)))  float  v8f;

__device__ __forceinline__ v8f wmma_bf16(v16bf a, v16bf b, v8f c) {
  // D(16x16,f32) = A(16x32,bf16) x B(32x16,bf16) + C
  return __builtin_amdgcn_wmma_f32_16x16x32_bf16(false, a, false, b, (short)0, c,
                                                 false, false);
}

// CDNA5 async global->LDS copy (16B per lane), tracked by ASYNCcnt.
// LDS operand = low 32 bits of the flat pointer (ISA 10.2: LDS aperture
// truncates addr[31:0] to the LDS byte offset).
__device__ __forceinline__ void async_cp16(const __bf16* gsrc, __bf16* ldst) {
  asm volatile("global_load_async_to_lds_b128 %0, %1, off"
               :
               : "v"((unsigned)(size_t)ldst), "v"(gsrc)
               : "memory");
}
__device__ __forceinline__ void wait_async_le(int) {}  // (doc helper)
__device__ __forceinline__ void wait_async4() {
  asm volatile("s_wait_asynccnt 0x4" ::: "memory");
}
__device__ __forceinline__ void wait_async0() {
  asm volatile("s_wait_asynccnt 0x0" ::: "memory");
}

// ---------------------------------------------------------------------------
// LayerNorm over E=1024, one block (256 thr, 8 waves) per row, bf16 output.
// ---------------------------------------------------------------------------
__global__ __launch_bounds__(256) void ln_kernel(const float* __restrict__ x,
                                                 const float* __restrict__ gam,
                                                 const float* __restrict__ bet,
                                                 __bf16* __restrict__ out) {
  __shared__ float red[8];
  const int row = blockIdx.x;
  const float* xr = x + (size_t)row * 1024;
  const int base = threadIdx.x * 4;
  float4 v = *(const float4*)(xr + base);
  float s = v.x + v.y + v.z + v.w;
#pragma unroll
  for (int off = 16; off > 0; off >>= 1) s += __shfl_xor(s, off, 32);
  const int lane = threadIdx.x & 31, wv = threadIdx.x >> 5;
  if (lane == 0) red[wv] = s;
  __syncthreads();
  float tot = red[0] + red[1] + red[2] + red[3] + red[4] + red[5] + red[6] + red[7];
  const float mean = tot * (1.0f / 1024.0f);
  const float d0 = v.x - mean, d1 = v.y - mean, d2 = v.z - mean, d3 = v.w - mean;
  float q = d0 * d0 + d1 * d1 + d2 * d2 + d3 * d3;
#pragma unroll
  for (int off = 16; off > 0; off >>= 1) q += __shfl_xor(q, off, 32);
  __syncthreads();
  if (lane == 0) red[wv] = q;
  __syncthreads();
  tot = red[0] + red[1] + red[2] + red[3] + red[4] + red[5] + red[6] + red[7];
  const float rstd = rsqrtf(tot * (1.0f / 1024.0f) + 1e-5f);
  float4 gv = *(const float4*)(gam + base);
  float4 bv = *(const float4*)(bet + base);
  __bf16* o = out + (size_t)row * 1024 + base;
  o[0] = (__bf16)(d0 * rstd * gv.x + bv.x);
  o[1] = (__bf16)(d1 * rstd * gv.y + bv.y);
  o[2] = (__bf16)(d2 * rstd * gv.z + bv.z);
  o[3] = (__bf16)(d3 * rstd * gv.w + bv.w);
}

// ---------------------------------------------------------------------------
// f32 -> bf16 conversion kernels (weights), pos_emb padded 513 -> 544 cols.
// ---------------------------------------------------------------------------
__global__ void cvt_kernel(const float* __restrict__ src, __bf16* __restrict__ dst,
                           int n) {
  int i = blockIdx.x * 256 + threadIdx.x;
  if (i < n) dst[i] = (__bf16)src[i];
}

__global__ void cvt_pos_kernel(const float* __restrict__ src,
                               __bf16* __restrict__ dst) {
  int i = blockIdx.x * 256 + threadIdx.x;
  if (i >= 64 * 544) return;
  int j = i % 544, k = i / 544;
  dst[i] = (__bf16)(j < 513 ? src[k * 513 + j] : 0.0f);
}

// ---------------------------------------------------------------------------
// bf16 WMMA GEMM: C[M,N] = act(A[M,K] * W[N,K]^T + bias) (+ resid)
// Block tile 128x128x32, 256 threads = 8 waves (4 along M x 2 along N).
// Double-buffered LDS filled with global_load_async_to_lds_b128; per step a
// thread issues 4 async 16B copies for tile i+1, waits asynccnt<=4 so tile i
// is resident while tile i+1 streams, then runs 2x4 v_wmma_f32_16x16x32_bf16.
// ---------------------------------------------------------------------------
__global__ __launch_bounds__(256) void gemm_kernel(
    const __bf16* __restrict__ A, const __bf16* __restrict__ W,
    const float* __restrict__ bias, const float* __restrict__ resid,
    float* __restrict__ Cf, __bf16* __restrict__ Cb, int M, int N, int K,
    int relu) {
  __shared__ __bf16 As[2][128][40];  // 80B rows (16B aligned)
  __shared__ __bf16 Ws[2][128][48];  // 96B rows (16B aligned), Ws[.][n][k]
  const int tid = threadIdx.x;
  const int lane = tid & 31, wave = tid >> 5;
  const int wm = wave & 3, wn = wave >> 2;
  const int h = lane >> 4, ln = lane & 15;
  const int m0 = blockIdx.x * 128, n0 = blockIdx.y * 128;

  v8f acc[2][4];
#pragma unroll
  for (int i = 0; i < 2; ++i)
#pragma unroll
    for (int j = 0; j < 4; ++j) acc[i][j] = (v8f)(0.0f);

  // staging map: tile = 128 rows x 32 cols bf16 = 512 x 16B chunks,
  // 2 chunks per thread (8 bf16 each)
  const int c0 = tid * 2, c1 = tid * 2 + 1;
  const int r0 = c0 >> 2, a0 = (c0 & 3) * 8;
  const int r1 = c1 >> 2, a1 = (c1 & 3) * 8;

  auto stage = [&](int buf, int k0) {
    async_cp16(A + (size_t)(m0 + r0) * K + k0 + a0, &As[buf][r0][a0]);
    async_cp16(A + (size_t)(m0 + r1) * K + k0 + a1, &As[buf][r1][a1]);
    async_cp16(W + (size_t)(n0 + r0) * K + k0 + a0, &Ws[buf][r0][a0]);
    async_cp16(W + (size_t)(n0 + r1) * K + k0 + a1, &Ws[buf][r1][a1]);
  };

  stage(0, 0);
  const int nsteps = K >> 5;
  for (int s = 0; s < nsteps; ++s) {
    const int buf = s & 1;
    if (s + 1 < nsteps) {
      stage(buf ^ 1, (s + 1) << 5);
      if (s + 2 < nsteps)  // cache hint two tiles ahead (global_prefetch_b8)
        __builtin_prefetch(A + (size_t)(m0 + r0) * K + ((s + 2) << 5), 0, 1);
      wait_async4();  // tile s complete (in-order), tile s+1 still in flight
    } else {
      wait_async0();
    }
    __syncthreads();

    v16bf amat[2], bmat[4];
#pragma unroll
    for (int mf = 0; mf < 2; ++mf) {
      // A 16x32 bf16 layout: elem e -> k = h*16+e
      const __bf16* p = &As[buf][wm * 32 + mf * 16 + ln][0];
      v8bf lo = *(const v8bf*)(p + h * 16);
      v8bf hi = *(const v8bf*)(p + h * 16 + 8);
#pragma unroll
      for (int e = 0; e < 8; ++e) { amat[mf][e] = lo[e]; amat[mf][8 + e] = hi[e]; }
    }
#pragma unroll
    for (int nf = 0; nf < 4; ++nf)  // B 32x16: elem e -> k = h*16+e, col = ln
      bmat[nf] = *(const v16bf*)&Ws[buf][wn * 64 + nf * 16 + ln][h * 16];

#pragma unroll
    for (int mf = 0; mf < 2; ++mf)
#pragma unroll
      for (int nf = 0; nf < 4; ++nf)
        acc[mf][nf] = wmma_bf16(amat[mf], bmat[nf], acc[mf][nf]);
    __syncthreads();  // safe to overwrite buf next iteration
  }

  // C/D layout: lane col n=ln, VGPR r -> row r + 8*h
#pragma unroll
  for (int mf = 0; mf < 2; ++mf) {
#pragma unroll
    for (int nf = 0; nf < 4; ++nf) {
      const int gc = n0 + wn * 64 + nf * 16 + ln;
      const float bb = bias ? bias[gc] : 0.0f;
#pragma unroll
      for (int r = 0; r < 8; ++r) {
        const int gr = m0 + wm * 32 + mf * 16 + r + 8 * h;
        float val = acc[mf][nf][r] + bb;
        if (relu) val = fmaxf(val, 0.0f);
        if (resid) val += resid[(size_t)gr * N + gc];
        if (Cf) Cf[(size_t)gr * N + gc] = val;
        if (Cb) Cb[(size_t)gr * N + gc] = (__bf16)val;
      }
    }
  }
}

// ---------------------------------------------------------------------------
// Banded attention, flash-style. One wave (32 thr) per (g, bh, 16-row s-tile).
// scores[s, t] = (q·k_win + qe[s, t-s]) * nk^-0.5, band 0 <= t-s <= 512,
// group 0 masks t < 512 (padding region). Online softmax, O = P x V.
// (Register staging kept here: it fuses f32->bf16 conversion + masked zero
// fill, which a raw async byte copy cannot do.)
// ---------------------------------------------------------------------------
__global__ __launch_bounds__(32) void attn_kernel(const float* __restrict__ qkv,
                                                  const __bf16* __restrict__ pos,
                                                  __bf16* __restrict__ aout) {
  __shared__ float qe[16][544];    // relative-position logits for this s-tile
  __shared__ __bf16 KsT[32][72];   // K tile transposed: [t_local][ck]
  __shared__ __bf16 VsT[64][40];   // V tile transposed: [cv][t_local]
  __shared__ __bf16 Ps[16][40];    // P tile (row-major) for A-frag gather

  const int bid = blockIdx.x;
  const int st = bid & 31;
  const int bh = (bid >> 5) & 31;
  const int g = bid >> 10;
  const int b = bh >> 4, hh = bh & 15;
  const int s0 = st * 16;
  const int lane = threadIdx.x;
  const int h = lane >> 4, ln = lane & 15;

  // --- q A-fragments (16 rows x K=64 -> two 16x32 fragments) ---
  v16bf qa[2];
  const size_t qrow = ((size_t)(g * 512 + s0 + ln) * 2 + b) * 3072 + hh * 192;
#pragma unroll
  for (int kc = 0; kc < 2; ++kc) {
#pragma unroll
    for (int e = 0; e < 16; ++e) {
      const int ck = kc * 32 + h * 16 + e;
      qa[kc][e] = (__bf16)qkv[qrow + ck];
    }
  }

  // --- qe[s, j] = sum_k q[s,k] * pos_emb[k, j] (WMMA, 33 j-tiles) ---
  for (int jt = 0; jt < 33; ++jt) {
    v8f c = (v8f)(0.0f);
#pragma unroll
    for (int kc = 0; kc < 2; ++kc) {
      v16bf pb;
#pragma unroll
      for (int e = 0; e < 16; ++e)
        pb[e] = pos[(size_t)(kc * 32 + h * 16 + e) * 544 + jt * 16 + ln];
      c = wmma_bf16(qa[kc], pb, c);
    }
#pragma unroll
    for (int r = 0; r < 8; ++r) qe[r + 8 * h][jt * 16 + ln] = c[r];
  }

  float rmax[8], rsum[8];
  v8f O[4];
#pragma unroll
  for (int r = 0; r < 8; ++r) { rmax[r] = -1e30f; rsum[r] = 0.0f; }
#pragma unroll
  for (int cv = 0; cv < 4; ++cv) O[cv] = (v8f)(0.0f);

  for (int step = 0; step < 17; ++step) {
    const int t0 = s0 + step * 32;              // window coordinate tile start
    if (g == 0 && t0 + 31 < 512) continue;      // fully inside padding mask

    // stage K/V tile (coalesced along channel, transposed into LDS)
    for (int i = lane; i < 2048; i += 32) {
      const int tl = i >> 6, ck = i & 63;
      const int tw = t0 + tl;
      float kvv = 0.0f, vvv = 0.0f;
      if (tw < 1024 && (g > 0 || tw >= 512)) {
        const size_t rb = ((size_t)((g - 1) * 512 + tw) * 2 + b) * 3072 + hh * 192;
        kvv = qkv[rb + 64 + ck];
        vvv = qkv[rb + 128 + ck];
      }
      KsT[tl][ck] = (__bf16)kvv;
      VsT[ck][tl] = (__bf16)vvv;
    }

    // --- two 16x16 score tiles, K=64 chained WMMA ---
    v8f S[2];
#pragma unroll
    for (int tt = 0; tt < 2; ++tt) {
      v8f c = (v8f)(0.0f);
#pragma unroll
      for (int kc = 0; kc < 2; ++kc) {
        v16bf kb = *(const v16bf*)&KsT[tt * 16 + ln][kc * 32 + h * 16];
        c = wmma_bf16(qa[kc], kb, c);
      }
      const int t = t0 + tt * 16 + ln;
#pragma unroll
      for (int r = 0; r < 8; ++r) {
        const int m = r + 8 * h;
        const int j = t - (s0 + m);
        float val;
        if (j < 0 || j > 512 || (g == 0 && t < 512)) val = -1e30f;
        else val = (c[r] + qe[m][j]) * 0.125f;  // nk^-0.5 = 1/8
        c[r] = val;
      }
      S[tt] = c;
    }

    // --- online softmax (rows live across 16 lanes of one half-wave) ---
    float nmax[8], alpha[8];
#pragma unroll
    for (int r = 0; r < 8; ++r) {
      float mx = fmaxf(S[0][r], S[1][r]);
#pragma unroll
      for (int o = 1; o < 16; o <<= 1) mx = fmaxf(mx, __shfl_xor(mx, o, 32));
      nmax[r] = fmaxf(rmax[r], mx);
      alpha[r] = __expf(rmax[r] - nmax[r]);
      rmax[r] = nmax[r];
    }
#pragma unroll
    for (int tt = 0; tt < 2; ++tt)
#pragma unroll
      for (int r = 0; r < 8; ++r) S[tt][r] = __expf(S[tt][r] - nmax[r]);
#pragma unroll
    for (int r = 0; r < 8; ++r) {
      float sm = S[0][r] + S[1][r];
#pragma unroll
      for (int o = 1; o < 16; o <<= 1) sm += __shfl_xor(sm, o, 32);
      rsum[r] = rsum[r] * alpha[r] + sm;
    }
#pragma unroll
    for (int cv = 0; cv < 4; ++cv)
#pragma unroll
      for (int r = 0; r < 8; ++r) O[cv][r] *= alpha[r];

    // --- P (C-layout) -> A-fragment via LDS transpose ---
#pragma unroll
    for (int tt = 0; tt < 2; ++tt)
#pragma unroll
      for (int r = 0; r < 8; ++r) Ps[r + 8 * h][tt * 16 + ln] = (__bf16)S[tt][r];
    v16bf pa;
    {
      const __bf16* pp = &Ps[ln][0];
      v8bf lo = *(const v8bf*)(pp + h * 16);
      v8bf hi = *(const v8bf*)(pp + h * 16 + 8);
#pragma unroll
      for (int e = 0; e < 8; ++e) { pa[e] = lo[e]; pa[8 + e] = hi[e]; }
    }
    // --- O(16x64) += P(16x32) x V(32x64) ---
#pragma unroll
    for (int cv = 0; cv < 4; ++cv) {
      v16bf vb = *(const v16bf*)&VsT[cv * 16 + ln][h * 16];
      O[cv] = wmma_bf16(pa, vb, O[cv]);
    }
  }

  // normalize and store (row r+8h, col cv*16+ln), as bf16 for the out-proj GEMM
#pragma unroll
  for (int cv = 0; cv < 4; ++cv) {
#pragma unroll
    for (int r = 0; r < 8; ++r) {
      const int m = r + 8 * h;
      const int t = g * 512 + s0 + m;
      aout[((size_t)t * 2 + b) * 1024 + hh * 64 + cv * 16 + ln] =
          (__bf16)(O[cv][r] / rsum[r]);
    }
  }
}

// ---------------------------------------------------------------------------
// nxt = kv_cat[-512:] : last group's k,v rows -> d_out tail (f32).
// ---------------------------------------------------------------------------
__global__ void copy_kv_kernel(const float* __restrict__ qkv,
                               float* __restrict__ dst) {
  int i = blockIdx.x * 256 + threadIdx.x;
  if (i >= 512 * 32 * 128) return;
  const int c = i & 127;
  const int bh = (i >> 7) & 31;
  const int s = i >> 12;
  const int b = bh >> 4, hh = bh & 15;
  const size_t t = 3584 + s;
  dst[i] = qkv[(t * 2 + b) * 3072 + hh * 192 + 64 + c];
}

// ---------------------------------------------------------------------------
extern "C" void kernel_launch(void* const* d_in, const int* in_sizes, int n_in,
                              void* d_out, int out_size, void* d_ws,
                              size_t ws_size, hipStream_t stream) {
  (void)in_sizes; (void)n_in; (void)out_size; (void)ws_size;
  const float* x      = (const float*)d_in[0];
  const float* ln1_g  = (const float*)d_in[1];
  const float* ln1_b  = (const float*)d_in[2];
  const float* qkv_w  = (const float*)d_in[3];
  const float* qkv_b  = (const float*)d_in[4];
  const float* posw   = (const float*)d_in[5];
  const float* out_w  = (const float*)d_in[6];
  const float* ln2_g  = (const float*)d_in[7];
  const float* ln2_b  = (const float*)d_in[8];
  const float* fc1_w  = (const float*)d_in[9];
  const float* fc1_b  = (const float*)d_in[10];
  const float* fc2_w  = (const float*)d_in[11];
  const float* fc2_b  = (const float*)d_in[12];

  char* ws = (char*)d_ws;
  size_t off = 0;
  auto take = [&](size_t bytes) -> void* {
    void* p = ws + off;
    off += (bytes + 255) & ~(size_t)255;
    return p;
  };
  const size_t R = 8192;  // T*B rows
  __bf16* hln1  = (__bf16*)take(R * 1024 * 2);
  __bf16* wqkvb = (__bf16*)take((size_t)3072 * 1024 * 2);
  __bf16* woutb = (__bf16*)take((size_t)1024 * 1024 * 2);
  __bf16* wfc1b = (__bf16*)take((size_t)4096 * 1024 * 2);
  __bf16* wfc2b = (__bf16*)take((size_t)1024 * 4096 * 2);
  __bf16* posb  = (__bf16*)take((size_t)64 * 544 * 2);
  float*  qkvf  = (float*)take(R * 3072 * 4);
  __bf16* attnb = (__bf16*)take(R * 1024 * 2);
  float*  out1f = (float*)take(R * 1024 * 4);
  __bf16* h2b   = (__bf16*)take(R * 1024 * 2);
  __bf16* ff1b  = (__bf16*)take(R * 4096 * 2);

  // 1) LN1
  ln_kernel<<<8192, 256, 0, stream>>>(x, ln1_g, ln1_b, hln1);
  // 2) weight + pos_emb conversions to bf16
  cvt_kernel<<<(3072 * 1024 + 255) / 256, 256, 0, stream>>>(qkv_w, wqkvb, 3072 * 1024);
  cvt_kernel<<<(1024 * 1024 + 255) / 256, 256, 0, stream>>>(out_w, woutb, 1024 * 1024);
  cvt_kernel<<<(4096 * 1024 + 255) / 256, 256, 0, stream>>>(fc1_w, wfc1b, 4096 * 1024);
  cvt_kernel<<<(1024 * 4096 + 255) / 256, 256, 0, stream>>>(fc2_w, wfc2b, 1024 * 4096);
  cvt_pos_kernel<<<(64 * 544 + 255) / 256, 256, 0, stream>>>(posw, posb);
  // 3) QKV projection (f32 out for attention + kv export)
  {
    dim3 grid(64, 24);
    gemm_kernel<<<grid, 256, 0, stream>>>(hln1, wqkvb, qkv_b, nullptr, qkvf,
                                          nullptr, 8192, 3072, 1024, 0);
  }
  // 4) banded attention: 8 groups x 32 (b,h) x 32 s-tiles
  attn_kernel<<<8192, 32, 0, stream>>>(qkvf, posb, attnb);
  // 5) out-projection + residual x -> out1 (f32)
  {
    dim3 grid(64, 8);
    gemm_kernel<<<grid, 256, 0, stream>>>(attnb, woutb, nullptr, x, out1f,
                                          nullptr, 8192, 1024, 1024, 0);
  }
  // 6) LN2
  ln_kernel<<<8192, 256, 0, stream>>>(out1f, ln2_g, ln2_b, h2b);
  // 7) FC1 + ReLU (bf16 out)
  {
    dim3 grid(64, 32);
    gemm_kernel<<<grid, 256, 0, stream>>>(h2b, wfc1b, fc1_b, nullptr, nullptr,
                                          ff1b, 8192, 4096, 1024, 1);
  }
  // 8) FC2 + residual out1 -> d_out (f32)
  {
    dim3 grid(64, 8);
    gemm_kernel<<<grid, 256, 0, stream>>>(ff1b, wfc2b, fc2_b, out1f,
                                          (float*)d_out, nullptr, 8192, 1024,
                                          4096, 0);
  }
  // 9) nxt = kv_cat[-512:]
  copy_kv_kernel<<<(512 * 32 * 128 + 255) / 256, 256, 0, stream>>>(
      qkvf, (float*)d_out + (size_t)8388608);
}
